// MultiHeadAttention_9680856285263
// MI455X (gfx1250) — matrix operand order repaired and checked by reading, not compiled, not run on hardware
//
#include <hip/hip_runtime.h>
#include <hip/hip_bf16.h>
#include <stdint.h>

typedef __attribute__((ext_vector_type(16))) _Float16 v16h;
typedef __attribute__((ext_vector_type(8)))  float    v8f;

// ---- gfx1250 async global->LDS path (ASYNCcnt) via inline asm -------------
// Builtin exists but its pointer params live in the unspellable __device__
// LangAS, so we issue the instruction directly. VDST = LDS byte offset
// (low 32 bits of generic LDS pointer), VADDR = 64-bit global address.
#if defined(__AMDGCN__)
#define ASYNC_LDS 1
#endif

__device__ inline void async_copy16(const _Float16* gsrc, _Float16* ldst) {
#ifdef ASYNC_LDS
    uint32_t lds = (uint32_t)(uintptr_t)ldst;
    uint64_t ga  = (uint64_t)(uintptr_t)gsrc;
    asm volatile("global_load_async_to_lds_b128 %0, %1, off"
                 :: "v"(lds), "v"(ga) : "memory");
#else
    *(uint4*)ldst = *(const uint4*)gsrc;
#endif
}
__device__ inline void async_join() {
#ifdef ASYNC_LDS
    asm volatile("s_wait_asynccnt 0x0" ::: "memory");
#endif
}

__device__ inline v8f wmma_f16(v16h a, v16h b, v8f c) {
    return __builtin_amdgcn_wmma_f32_16x16x32_f16(
        false, a, false, b, (short)0, c, false, false);
}

union FragH { v16h h; uint32_t u[8]; };

// A-fragment: 16x32 f16 row-major (ld elems/row).
// lane m=lane%16, g=lane/16; VGPR j<4 -> K=g*8+2j, j>=4 -> K=16+g*8+2(j-4).
__device__ inline v16h load_frag_a(const _Float16* p, int ld) {
    const int lane = threadIdx.x & 31;
    const int m = lane & 15, g = lane >> 4;
    const uint32_t* p32 = (const uint32_t*)p;
    FragH a;
#pragma unroll
    for (int j = 0; j < 8; ++j) {
        int k = (j < 4) ? (g * 8 + 2 * j) : (16 + g * 8 + 2 * (j - 4));
        a.u[j] = p32[(m * ld + k) >> 1];
    }
    return a.h;
}

// B-fragment: 32x16 f16 from [n][k] (transposed) buffer, ld = k-stride.
// lane n=lane%16, g; VGPR j<4 -> K=g*8+2j, j>=4 -> K=16+g*8+2(j-4).
__device__ inline v16h load_frag_bt(const _Float16* p, int ld) {
    const int lane = threadIdx.x & 31;
    const int n = lane & 15, g = lane >> 4;
    const uint32_t* p32 = (const uint32_t*)p;
    FragH b;
#pragma unroll
    for (int j = 0; j < 8; ++j) {
        int k = (j < 4) ? (g * 8 + 2 * j) : (16 + g * 8 + 2 * (j - 4));
        b.u[j] = p32[(n * ld + k) >> 1];
    }
    return b.h;
}

// ---------------------------------------------------------------- converts
__global__ void cvt_f32_to_f16_k(const float* __restrict__ in,
                                 _Float16* __restrict__ out, int n) {
    int i = blockIdx.x * blockDim.x + threadIdx.x;
    int stride = gridDim.x * blockDim.x;
    for (; i < n; i += stride) out[i] = (_Float16)in[i];
}

// ---------------------------------------------------------------- GEMM core
// Block tile 128(M) x 64(N), K-step 32, 256 threads = 8 waves.
// Wave w owns a 16x64 strip: 1 A-frag feeds 4 WMMAs per K-step.
// Ping-pong LDS; async path issues ASYNC loads ahead of compute, sync path
// register-pipelines (global loads before WMMA, ds stores after).
#define GBM 128
#define GBN 64
#define GBK 32

template <int KDIM, int NTOT>
__device__ inline void gemm_core(const _Float16* __restrict__ A,
                                 const _Float16* __restrict__ Bw,
                                 _Float16* As, _Float16* Bst,
                                 int m0, int n0, v8f acc[4]) {
    const int t = threadIdx.x;
    const int wave = t >> 5;
    const int arow0 = (t + 0)   >> 2, apart0 = (t + 0)   & 3;
    const int arow1 = (t + 256) >> 2, apart1 = (t + 256) & 3;
    const int bkr = t >> 3, bpart = t & 7;

    // prologue: stage tile 0 directly
    {
        const _Float16* s0 = A + (size_t)(m0 + arow0) * KDIM + apart0 * 8;
        const _Float16* s1 = A + (size_t)(m0 + arow1) * KDIM + apart1 * 8;
        async_copy16(s0, &As[arow0 * GBK + apart0 * 8]);
        async_copy16(s1, &As[arow1 * GBK + apart1 * 8]);
        uint4 d = *(const uint4*)(Bw + (size_t)bkr * NTOT + n0 + bpart * 8);
        const _Float16* dh = (const _Float16*)&d;
#pragma unroll
        for (int e = 0; e < 8; ++e) Bst[(bpart * 8 + e) * GBK + bkr] = dh[e];
    }
    async_join();
    __syncthreads();

    for (int k0 = 0; k0 < KDIM; k0 += GBK) {
        const int cur = (k0 / GBK) & 1, nxt = cur ^ 1;
        const bool has_next = (k0 + GBK) < KDIM;
        _Float16* Asn = &As[nxt * (GBM * GBK)];
        _Float16* Bsn = &Bst[nxt * (GBN * GBK)];
#ifdef ASYNC_LDS
        if (has_next) {
            async_copy16(A + (size_t)(m0 + arow0) * KDIM + (k0 + GBK) + apart0 * 8,
                         &Asn[arow0 * GBK + apart0 * 8]);
            async_copy16(A + (size_t)(m0 + arow1) * KDIM + (k0 + GBK) + apart1 * 8,
                         &Asn[arow1 * GBK + apart1 * 8]);
        }
        uint4 rb;
        if (has_next)
            rb = *(const uint4*)(Bw + (size_t)(k0 + GBK + bkr) * NTOT + n0 + bpart * 8);
#else
        uint4 ra0, ra1, rb;
        if (has_next) {
            ra0 = *(const uint4*)(A + (size_t)(m0 + arow0) * KDIM + (k0 + GBK) + apart0 * 8);
            ra1 = *(const uint4*)(A + (size_t)(m0 + arow1) * KDIM + (k0 + GBK) + apart1 * 8);
            rb  = *(const uint4*)(Bw + (size_t)(k0 + GBK + bkr) * NTOT + n0 + bpart * 8);
        }
#endif
        if (k0 + 2 * GBK < KDIM)
            __builtin_prefetch(A + (size_t)(m0 + arow0) * KDIM + k0 + 2 * GBK, 0, 1);

        v16h a = load_frag_a(&As[cur * (GBM * GBK) + wave * 16 * GBK], GBK);
#pragma unroll
        for (int c = 0; c < 4; ++c) {
            v16h b = load_frag_bt(&Bst[cur * (GBN * GBK) + c * 16 * GBK], GBK);
            acc[c] = wmma_f16(a, b, acc[c]);
        }

        if (has_next) {
#ifndef ASYNC_LDS
            *(uint4*)(&Asn[arow0 * GBK + apart0 * 8]) = ra0;
            *(uint4*)(&Asn[arow1 * GBK + apart1 * 8]) = ra1;
#endif
            const _Float16* dh = (const _Float16*)&rb;
#pragma unroll
            for (int e = 0; e < 8; ++e) Bsn[(bpart * 8 + e) * GBK + bkr] = dh[e];
        }
        async_join();
        __syncthreads();
    }
}

// ---------------------------------------------------------------- QKV GEMM
__global__ __launch_bounds__(256)
void qkv_gemm_k(const _Float16* __restrict__ A, const _Float16* __restrict__ Bw,
                _Float16* __restrict__ Q, _Float16* __restrict__ Kt,
                _Float16* __restrict__ Vt) {
    __shared__ __align__(16) _Float16 As[2 * GBM * GBK];
    __shared__ __align__(16) _Float16 Bst[2 * GBN * GBK];
    const int m0 = blockIdx.y * GBM, n0 = blockIdx.x * GBN;
    const int t = threadIdx.x, wave = t >> 5;
    const int lane = t & 31, n = lane & 15, g = lane >> 4;
    v8f acc[4] = {};
    gemm_core<768, 2304>(A, Bw, As, Bst, m0, n0, acc);

#pragma unroll
    for (int c = 0; c < 4; ++c) {
        int col = n0 + c * 16 + n;                 // 0..2303
        int which = col / 768, cc = col % 768;
        int head = cc >> 6, d = cc & 63;
        _Float16* dst = (which == 0) ? Q : ((which == 1) ? Kt : Vt);
        float scl = (which == 0) ? 0.125f : 1.0f;  // hd^-0.5
#pragma unroll
        for (int r = 0; r < 8; ++r) {
            int row = m0 + wave * 16 + r + 8 * g;
            int b = row >> 10, nq = row & 1023;
            dst[(((size_t)(b * 12 + head)) * 1024 + nq) * 64 + d] =
                (_Float16)(acc[c][r] * scl);
        }
    }
}

// ---------------------------------------------------------------- flash attention
__global__ __launch_bounds__(128)
void flash_attn_k(const _Float16* __restrict__ Q, const _Float16* __restrict__ Kt,
                  const _Float16* __restrict__ Vt,
                  const float* __restrict__ bias_table,
                  _Float16* __restrict__ Oh) {
    constexpr int Nseq = 1024, HD = 64;
    __shared__ __align__(16) _Float16 Ks[32 * 64];      // [key][dim]
    __shared__ __align__(16) _Float16 Vs[64 * 32];      // [dim][key]
    __shared__ __align__(16) _Float16 Ps[4 * 16 * 32];  // per-wave P [row][key]
    const int t = threadIdx.x;
    const int wave = t >> 5, lane = t & 31;
    const int n = lane & 15, g = lane >> 4;
    const int bh = blockIdx.y;
    const int h = bh % 12;
    const int q0 = blockIdx.x * 64 + wave * 16;

    const _Float16* qbase = Q + ((size_t)bh * Nseq + q0) * HD;
    v16h qf0 = load_frag_a(qbase, HD);
    v16h qf1 = load_frag_a(qbase + 32, HD);

    v8f o0 = {}, o1 = {}, o2 = {}, o3 = {};
    float mrow[8], lrow[8];
#pragma unroll
    for (int r = 0; r < 8; ++r) { mrow[r] = -1e30f; lrow[r] = 0.0f; }

    for (int kb = 0; kb < Nseq; kb += 32) {
        __syncthreads();
        {   // stage K (async identity copy) and V (manual transpose)
            const _Float16* kg = Kt + ((size_t)bh * Nseq + kb) * HD;
            const _Float16* vg = Vt + ((size_t)bh * Nseq + kb) * HD;
#pragma unroll
            for (int c = 0; c < 2; ++c) {
                int chunk = t + c * 128;
                int kr = chunk >> 3, part = chunk & 7;
                async_copy16(kg + kr * HD + part * 8, &Ks[kr * 64 + part * 8]);
                uint4 dv = *(const uint4*)(vg + kr * HD + part * 8);
                const _Float16* dvh = (const _Float16*)&dv;
#pragma unroll
                for (int e = 0; e < 8; ++e) Vs[(part * 8 + e) * 32 + kr] = dvh[e];
            }
        }
        async_join();
        __syncthreads();

        // S = (Q*scale) @ K^T : two 16x16 key tiles, K-dim 64 -> 2 wmma each
        v8f s0 = {}, s1 = {};
        {
            v16h b00 = load_frag_bt(&Ks[0 * 16 * 64 + 0],  64);
            v16h b01 = load_frag_bt(&Ks[0 * 16 * 64 + 32], 64);
            v16h b10 = load_frag_bt(&Ks[1 * 16 * 64 + 0],  64);
            v16h b11 = load_frag_bt(&Ks[1 * 16 * 64 + 32], 64);
            s0 = wmma_f16(qf0, b00, s0);
            s0 = wmma_f16(qf1, b01, s0);
            s1 = wmma_f16(qf0, b10, s1);
            s1 = wmma_f16(qf1, b11, s1);
        }

        // + Swin relative position bias (L2-resident gather)
#pragma unroll
        for (int tt = 0; tt < 2; ++tt) {
            v8f* s = tt ? &s1 : &s0;
            int kcol = kb + tt * 16 + n;
            int kh = kcol >> 5, kw = kcol & 31;
#pragma unroll
            for (int r = 0; r < 8; ++r) {
                int qi = q0 + r + 8 * g;
                int rel = ((qi >> 5) - kh + 31) * 63 + ((qi & 31) - kw + 31);
                (*s)[r] += bias_table[rel * 12 + h];
            }
        }

        // online softmax (rows live in 16-lane groups -> xor-shuffle reductions)
        float alpha[8];
#pragma unroll
        for (int r = 0; r < 8; ++r) {
            float v = fmaxf(s0[r], s1[r]);
#pragma unroll
            for (int msk = 1; msk < 16; msk <<= 1) v = fmaxf(v, __shfl_xor(v, msk, 32));
            float mnew = fmaxf(mrow[r], v);
            alpha[r] = __expf(mrow[r] - mnew);
            mrow[r] = mnew;
            float p0 = __expf(s0[r] - mnew);
            float p1 = __expf(s1[r] - mnew);
            s0[r] = p0; s1[r] = p1;
            float rs = p0 + p1;
#pragma unroll
            for (int msk = 1; msk < 16; msk <<= 1) rs += __shfl_xor(rs, msk, 32);
            lrow[r] = lrow[r] * alpha[r] + rs;
        }

        // P (C-layout f32) -> per-wave LDS [row][key] f16 -> reload in A-layout
        _Float16* pw = &Ps[wave * 16 * 32];
#pragma unroll
        for (int r = 0; r < 8; ++r) {
            int row = r + 8 * g;
            pw[row * 32 + n]      = (_Float16)s0[r];
            pw[row * 32 + 16 + n] = (_Float16)s1[r];
        }
        __syncthreads();

        v16h pa  = load_frag_a(pw, 32);
        v16h vb0 = load_frag_bt(&Vs[(0 * 16) * 32], 32);
        v16h vb1 = load_frag_bt(&Vs[(1 * 16) * 32], 32);
        v16h vb2 = load_frag_bt(&Vs[(2 * 16) * 32], 32);
        v16h vb3 = load_frag_bt(&Vs[(3 * 16) * 32], 32);
#pragma unroll
        for (int r = 0; r < 8; ++r) {
            o0[r] *= alpha[r]; o1[r] *= alpha[r];
            o2[r] *= alpha[r]; o3[r] *= alpha[r];
        }
        o0 = wmma_f16(pa, vb0, o0);
        o1 = wmma_f16(pa, vb1, o1);
        o2 = wmma_f16(pa, vb2, o2);
        o3 = wmma_f16(pa, vb3, o3);
    }

#pragma unroll
    for (int r = 0; r < 8; ++r) {
        float inv = 1.0f / lrow[r];
        int qrow = q0 + r + 8 * g;
        size_t base = ((size_t)(bh / 12) * 1024 + qrow) * 768 + h * 64;
        Oh[base + 0 * 16 + n] = (_Float16)(o0[r] * inv);
        Oh[base + 1 * 16 + n] = (_Float16)(o1[r] * inv);
        Oh[base + 2 * 16 + n] = (_Float16)(o2[r] * inv);
        Oh[base + 3 * 16 + n] = (_Float16)(o3[r] * inv);
    }
}

// ---------------------------------------------------------------- output projection
__global__ __launch_bounds__(256)
void proj_gemm_k(const _Float16* __restrict__ A, const _Float16* __restrict__ Bw,
                 const float* __restrict__ bias, float* __restrict__ out) {
    __shared__ __align__(16) _Float16 As[2 * GBM * GBK];
    __shared__ __align__(16) _Float16 Bst[2 * GBN * GBK];
    const int m0 = blockIdx.y * GBM, n0 = blockIdx.x * GBN;
    const int t = threadIdx.x, wave = t >> 5;
    const int lane = t & 31, n = lane & 15, g = lane >> 4;
    v8f acc[4] = {};
    gemm_core<768, 768>(A, Bw, As, Bst, m0, n0, acc);

#pragma unroll
    for (int c = 0; c < 4; ++c) {
        int col = n0 + c * 16 + n;
        float bb = bias[col];
#pragma unroll
        for (int r = 0; r < 8; ++r) {
            int row = m0 + wave * 16 + r + 8 * g;
            out[(size_t)row * 768 + col] = acc[c][r] + bb;
        }
    }
}

// ---------------------------------------------------------------- launcher
extern "C" void kernel_launch(void* const* d_in, const int* in_sizes, int n_in,
                              void* d_out, int out_size, void* d_ws, size_t ws_size,
                              hipStream_t stream) {
    (void)in_sizes; (void)n_in; (void)out_size; (void)ws_size;
    const float* x          = (const float*)d_in[0];   // (8,1024,768)
    const float* qkv_w      = (const float*)d_in[1];   // (768,2304)
    const float* proj_w     = (const float*)d_in[2];   // (768,768)
    const float* proj_b     = (const float*)d_in[3];   // (768,)
    const float* bias_table = (const float*)d_in[4];   // (65025,12)

    char* ws = (char*)d_ws;
    _Float16* xh  = (_Float16*)(ws + 0);          // 8192*768 f16
    _Float16* wqh = (_Float16*)(ws + 12582912);   // 768*2304
    _Float16* pwh = (_Float16*)(ws + 16121856);   // 768*768
    _Float16* q   = (_Float16*)(ws + 17301504);   // (8,12,1024,64)
    _Float16* k   = (_Float16*)(ws + 29884416);
    _Float16* v   = (_Float16*)(ws + 42467328);
    _Float16* ah  = (_Float16*)(ws + 55050240);   // (8,1024,768) f16

    cvt_f32_to_f16_k<<<dim3(1024), dim3(256), 0, stream>>>(x, xh, 8192 * 768);
    cvt_f32_to_f16_k<<<dim3(1024), dim3(256), 0, stream>>>(qkv_w, wqh, 768 * 2304);
    cvt_f32_to_f16_k<<<dim3(512),  dim3(256), 0, stream>>>(proj_w, pwh, 768 * 768);

    qkv_gemm_k<<<dim3(2304 / GBN, 8192 / GBM), 256, 0, stream>>>(xh, wqh, q, k, v);
    flash_attn_k<<<dim3(1024 / 64, 8 * 12), 128, 0, stream>>>(q, k, v, bias_table, ah);
    proj_gemm_k<<<dim3(768 / GBN, 8192 / GBM), 256, 0, stream>>>(ah, pwh, proj_b,
                                                                 (float*)d_out);
}
